// DynamicsSolver_3143916061062
// MI455X (gfx1250) — matrix operand order repaired and checked by reading, not compiled, not run on hardware
//
#include <hip/hip_runtime.h>

typedef _Float16 f16;
typedef _Float16 v16h __attribute__((ext_vector_type(16)));
typedef _Float16 v8h  __attribute__((ext_vector_type(8)));
typedef float    v8f  __attribute__((ext_vector_type(8)));
typedef float    v4f  __attribute__((ext_vector_type(4)));

#define DT_C    0.01f
#define BETA_C  0.25f
#define GAM_C   0.5f

__device__ __forceinline__ void lds_fence() { asm volatile("s_wait_dscnt 0" ::: "memory"); }

// ---- CDNA5 async global->LDS copy (ASYNCcnt-tracked) ----
__device__ __forceinline__ void async_copy_b128(void* lds_dst, const void* gsrc) {
  unsigned loff = (unsigned)(unsigned long long)lds_dst;   // low 32 bits of flat LDS addr = LDS offset
  asm volatile("global_load_async_to_lds_b128 %0, %1, off" :: "v"(loff), "v"(gsrc) : "memory");
}
__device__ __forceinline__ void async_wait0() { asm volatile("s_wait_asynccnt 0x0" ::: "memory"); }

// Stage a 128x128 f16 weight matrix (32KB) into LDS, whole workgroup cooperating.
__device__ __forceinline__ void stage_w128(f16* wlds, const f16* wglob, int tid) {
  __syncthreads();                    // previous consumers of wlds are done
  for (int i = tid; i < 2048; i += 128)
    async_copy_b128((char*)wlds + i * 16, (const char*)wglob + i * 16);
  async_wait0();
  __syncthreads();                    // all waves' copies landed
}

// ---------------- WMMA fragment helpers (CDNA5 16x16x32 f16 layouts) ----------------
__device__ __forceinline__ v16h a_frag(const f16* aRow, int kt, int hi) {
  union { v16h v; v8h h[2]; } u;
  const f16* p = aRow + kt * 32 + 16 * hi;
  u.h[0] = *(const v8h*)(p);
  u.h[1] = *(const v8h*)(p + 8);
  return u.v;
}

// GEMM: 16 rows x 128 outputs.  Weights f16 transposed+padded WT[n][k], row stride KT*32.
// For KT<=4 the A fragments are preloaded into registers and reused across all 8 N-tiles.
template <int KT>
__device__ __forceinline__ void gemm_128(const f16* aRow, int hi, int lane,
                                         const f16* wt, const float* bias, bool relu,
                                         f16* dst16, int ld16, float* dstf, int ldf) {
  const int N = lane & 15;
  v16h af[KT <= 4 ? KT : 1];
  if constexpr (KT <= 4) {
#pragma unroll
    for (int kt = 0; kt < KT; ++kt) af[kt] = a_frag(aRow, kt, hi);
  }
  for (int nt = 0; nt < 8; ++nt) {
    v8f acc = {0.f, 0.f, 0.f, 0.f, 0.f, 0.f, 0.f, 0.f};
    const f16* wrow = wt + (size_t)(nt * 16 + N) * (KT * 32);
#pragma unroll
    for (int kt = 0; kt < KT; ++kt) {
      v16h a;
      if constexpr (KT <= 4) a = af[kt]; else a = a_frag(aRow, kt, hi);
      v16h b = *(const v16h*)(wrow + kt * 32 + 16 * hi); // one 32-byte load: 16 contiguous k
      acc = __builtin_amdgcn_wmma_f32_16x16x32_f16(false, a, false, b, (short)0, acc, false, false);
    }
    float bN = bias[nt * 16 + N];
#pragma unroll
    for (int v = 0; v < 8; ++v) {
      int row = v + 8 * hi;
      float x = acc[v] + bN;
      if (dst16) {
        if (relu) x = fmaxf(x, 0.f);
        dst16[row * ld16 + nt * 16 + N] = (f16)x;
      } else {
        dstf[row * ldf + nt * 16 + N] = x;
      }
    }
  }
}

// GEMM: 16 rows x <=16 outputs (decoder heads, N padded to 16). Raw f32 out.
template <int KT>
__device__ __forceinline__ void gemm_16o(const f16* aRow, int hi, int lane,
                                         const f16* wt, const float* bias, int nb,
                                         float* dstf, int ldf) {
  const int N = lane & 15;
  v8f acc = {0.f, 0.f, 0.f, 0.f, 0.f, 0.f, 0.f, 0.f};
  const f16* wrow = wt + (size_t)N * (KT * 32);
#pragma unroll
  for (int kt = 0; kt < KT; ++kt) {
    v16h a = a_frag(aRow, kt, hi);
    v16h b = *(const v16h*)(wrow + kt * 32 + 16 * hi);
    acc = __builtin_amdgcn_wmma_f32_16x16x32_f16(false, a, false, b, (short)0, acc, false, false);
  }
  float bN = (N < nb) ? bias[N] : 0.f;
#pragma unroll
  for (int v = 0; v < 8; ++v) dstf[(v + 8 * hi) * ldf + N] = acc[v] + bN;
}

// Full-wave LayerNorm over 128 features: row = lane&15, each half-wave covers 64 cols.
// Partial sums combined across halves via shfl_xor(16); packed v8h output stores.
__device__ __forceinline__ void ln_rows(const float* scr, int lds_, const float* g, const float* bb,
                                        f16* dst, int ldd, int accum, int lane) {
  const int row = lane & 15, half = lane >> 4;
  const float* r = scr + row * lds_ + half * 64;
  float s = 0.f, s2 = 0.f;
#pragma unroll
  for (int c = 0; c < 16; ++c) {
    v4f x = *(const v4f*)(r + c * 4);
    s  += x[0] + x[1] + x[2] + x[3];
    s2 += x[0]*x[0] + x[1]*x[1] + x[2]*x[2] + x[3]*x[3];
  }
  s  += __shfl_xor(s, 16, 32);
  s2 += __shfl_xor(s2, 16, 32);
  float mu  = s * (1.f / 128.f);
  float var = fmaxf(s2 * (1.f / 128.f) - mu * mu, 0.f);
  float rs  = rsqrtf(var + 1e-5f);
  f16* d = dst + row * ldd + half * 64;
  const float* gg = g  + half * 64;
  const float* bp = bb + half * 64;
#pragma unroll
  for (int c = 0; c < 8; ++c) {
    v8h old = {};
    if (accum) old = *(const v8h*)(d + c * 8);
    v8h o;
#pragma unroll
    for (int j = 0; j < 8; ++j) {
      float y = (r[c*8+j] - mu) * rs * gg[c*8+j] + bp[c*8+j];
      if (accum) y += (float)old[j];
      o[j] = (f16)y;
    }
    *(v8h*)(d + c * 8) = o;
  }
}

// ---------------- utility kernels ----------------
__global__ void conv_w_kernel(const float* src, f16* dst, int K, int N, int Kp, int Np) {
  int i = blockIdx.x * 256 + threadIdx.x;
  if (i >= Kp * Np) return;
  int n = i / Kp, k = i - n * Kp;
  float v = (k < K && n < N) ? src[(size_t)k * N + n] : 0.f;
  dst[i] = (f16)v;
}

__global__ void zero_kernel(float* p, int n) {
  int i = blockIdx.x * 256 + threadIdx.x;
  if (i < n) p[i] = 0.f;
}

__global__ void prep_kernel(const float* pos, const float* vel, const float* pvel,
                            float* acc, float* x1, float* a1, float* v1, float* totres, int N) {
  int n = blockIdx.x * 256 + threadIdx.x;
  if (n >= N) return;
  totres[n] = 0.f;
  for (int j = 0; j < 3; ++j) {
    int i = n * 3 + j;
    float a  = (vel[i] - pvel[i]) / DT_C;
    float x  = pos[i] + vel[i] * DT_C + (0.5f - BETA_C) * DT_C * DT_C * a;
    float aa = (x - pos[i] - DT_C * vel[i]) / (BETA_C * DT_C * DT_C) - (1.f / (2.f * BETA_C) - 1.f) * a;
    float vv = vel[i] + DT_C * ((1.f - GAM_C) * a + GAM_C * aa);
    acc[i] = a; x1[i] = x; a1[i] = aa; v1[i] = vv;
  }
}

__global__ void final_kernel(const float* x1, const float* v1, const float* totres, float* out, int N) {
  int n = blockIdx.x * 256 + threadIdx.x;
  if (n >= N) return;
  out[n * 7 + 0] = x1[n * 3 + 0]; out[n * 7 + 1] = x1[n * 3 + 1]; out[n * 7 + 2] = x1[n * 3 + 2];
  out[n * 7 + 3] = v1[n * 3 + 0]; out[n * 7 + 4] = v1[n * 3 + 1]; out[n * 7 + 5] = v1[n * 3 + 2];
  out[n * 7 + 6] = totres[n];
}

// ---------------- node encoder: [type, |vel|] -> 128 latent ----------------
#define NE_WAVE_BYTES (16*32*2 + 16*128*2 + 16*132*4)
__global__ void node_enc_kernel(const f16* w0, const f16* w1,
                                const float* b0, const float* b1, const float* g, const float* be,
                                const int* node_type, const float* vel, f16* nlat, int N) {
  extern __shared__ char sm[];
  const int lane = threadIdx.x & 31, wave = threadIdx.x >> 5;
  char* wb = sm + (size_t)wave * NE_WAVE_BYTES;
  f16*  inb = (f16*)wb;
  f16*  hid = (f16*)(wb + 16*32*2);
  float* scr = (float*)(wb + 16*32*2 + 16*128*2);
  const int hi = lane >> 4, M = lane & 15;
  const int base = (blockIdx.x * 4 + wave) * 16;
  for (int i = lane; i < 512; i += 32) inb[i] = (f16)0.f;
  lds_fence();
  if (lane < 16) {
    int n = min(base + lane, N - 1);
    float t  = (float)node_type[n];
    float vx = vel[n*3], vy = vel[n*3+1], vz = vel[n*3+2];
    inb[lane * 32 + 0] = (f16)t;
    inb[lane * 32 + 1] = (f16)sqrtf(vx*vx + vy*vy + vz*vz);
  }
  lds_fence();
  gemm_128<1>(inb + M * 32, hi, lane, w0, b0, true, hid, 128, nullptr, 0);
  lds_fence();
  gemm_128<4>(hid + M * 128, hi, lane, w1, b1, false, nullptr, 0, scr, 132);
  lds_fence();
  ln_rows(scr, 132, g, be, hid, 128, 0, lane);
  lds_fence();
  {
    int n = base + M;
    if (n < N) {
      f16* dp = nlat + (size_t)n * 128 + hi * 64;
      const f16* sp = hid + M * 128 + hi * 64;
#pragma unroll
      for (int c = 0; c < 8; ++c) *(v8h*)(dp + c * 8) = *(const v8h*)(sp + c * 8);
    }
  }
}

// ---------------- fused edge block (the hot kernel) ----------------
struct EdgeParams {
  const f16 *ee0, *ee1, *ef0, *ef1, *ie0, *ie1, *i10, *i11;
  const float *ee_b0, *ee_b1, *ee_g, *ee_be;
  const float *ef_b0, *ef_b1, *ef_g, *ef_be;
  const float *ie_b0, *ie_b1, *ie_g, *ie_be;
  const float *i1_b0, *i1_b1;
  const float *lng, *lnb;
};

#define EDGE_WAVE_BYTES (16*384*2 + 16*128*2 + 16*32*2 + 16*132*4)  // 25856
#define EDGE_SMEM (4*EDGE_WAVE_BYTES + 128*128*2)                   // + 32KB staged weights
__global__ void edge_block_kernel(EdgeParams P,
                                  const int* senders, const int* receivers, const float* edge_attr,
                                  const float* x1, const float* v1, const float* vel,
                                  const float* vs_p, const float* dmin_p, const float* dmax_p,
                                  const f16* nlat, f16* hist, int use_hist,
                                  float* out_fij, int nE) {
  extern __shared__ char sm[];
  const int tid = threadIdx.x;
  const int lane = tid & 31, wave = tid >> 5;
  char* wb = sm + (size_t)wave * EDGE_WAVE_BYTES;
  f16*  inter = (f16*)wb;                                     // [16][384]: sl+rl | node_sum | edge_latent
  f16*  hid   = (f16*)(wb + 16*384*2);                        // [16][128]
  f16*  inb   = (f16*)(wb + 16*384*2 + 16*128*2);             // [16][32]
  float* scr  = (float*)(wb + 16*384*2 + 16*128*2 + 16*32*2); // [16][132] (also async gather buf)
  f16*  wlds  = (f16*)(sm + 4*EDGE_WAVE_BYTES);               // staged 128x128 weights (shared)
  const int hi = lane >> 4, M = lane & 15;
  const int e0 = (blockIdx.x * 4 + wave) * 16;

  float vax=0,vay=0,vaz=0, vbx=0,vby=0,vbz=0, vcx=0,vcy=0,vcz=0;
  float sfv[6], rfv[6];
  float edn = 0.f, ea = 0.f;
  int recv = 0, valid = 0;

  if (lane < 16) {
    int e = min(e0 + lane, nE - 1);
    valid = (e0 + lane) < nE;
    int s = senders[e], r = receivers[e];
    recv = r;
    float vs = vs_p[0] + 1e-8f;
    float spx=x1[s*3], spy=x1[s*3+1], spz=x1[s*3+2];
    float rpx=x1[r*3], rpy=x1[r*3+1], rpz=x1[r*3+2];
    float svx=v1[s*3]/vs, svy=v1[s*3+1]/vs, svz=v1[s*3+2]/vs;
    float rvx=v1[r*3]/vs, rvy=v1[r*3+1]/vs, rvz=v1[r*3+2]/vs;
    float smx=vel[s*3]/vs, smy=vel[s*3+1]/vs, smz=vel[s*3+2]/vs;
    float rmx=vel[r*3]/vs, rmy=vel[r*3+1]/vs, rmz=vel[r*3+2]/vs;
    float relx=rpx-spx, rely=rpy-spy, relz=rpz-spz;
    float nrm = sqrtf(relx*relx + rely*rely + relz*relz);
    float dist = fmaxf(nrm, 1e-6f);
    vax = relx/dist; vay = rely/dist; vaz = relz/dist;
    float dx=rvx-svx, dy=rvy-svy, dz=rvz-svz;
    float cx = dy*vaz - dz*vay, cy = dz*vax - dx*vaz, cz = dx*vay - dy*vax;
    float cl = fmaxf(sqrtf(cx*cx+cy*cy+cz*cz), 1e-6f);
    float bax=cx/cl, bay=cy/cl, baz=cz/cl;
    float sx=svx+rvx, sy=svy+rvy, sz=svz+rvz;
    float sl2 = fmaxf(sqrtf(sx*sx+sy*sy+sz*sz), 1e-6f);
    float bcx=sx/sl2, bcy=sy/sl2, bcz=sz/sl2;
    float bx=bax+bcx, by=bay+bcy, bz=baz+bcz;
    float bdva = bx*vax + by*vay + bz*vaz;
    float plx=bdva*vax, ply=bdva*vay, plz=bdva*vaz;
    float ppx=bx-plx, ppy=by-ply, ppz=bz-plz;
    float c2x = ppy*vaz - ppz*vay, c2y = ppz*vax - ppx*vaz, c2z = ppx*vay - ppy*vax;
    float l2 = fmaxf(sqrtf(c2x*c2x+c2y*c2y+c2z*c2z), 1e-6f);
    vbx=c2x/l2; vby=c2y/l2; vbz=c2z/l2;
    float c3x = ply*vbz - plz*vby, c3y = plz*vbx - plx*vbz, c3z = plx*vby - ply*vbx;
    float l3 = fmaxf(sqrtf(c3x*c3x+c3y*c3y+c3z*c3z), 1e-6f);
    vcx=c3x/l3; vcy=c3y/l3; vcz=c3z/l3;
    sfv[0]=vax*svx+vay*svy+vaz*svz; sfv[1]=vbx*svx+vby*svy+vbz*svz; sfv[2]=vcx*svx+vcy*svy+vcz*svz;
    sfv[3]=vax*smx+vay*smy+vaz*smz; sfv[4]=vbx*smx+vby*smy+vbz*smz; sfv[5]=vcx*smx+vcy*smy+vcz*smz;
    rfv[0]=-(vax*rvx+vay*rvy+vaz*rvz); rfv[1]=-(vbx*rvx+vby*rvy+vbz*rvz); rfv[2]=-(vcx*rvx+vcy*rvy+vcz*rvz);
    rfv[3]=-(vax*rmx+vay*rmy+vaz*rmz); rfv[4]=-(vbx*rmx+vby*rmy+vbz*rmz); rfv[5]=-(vcx*rmx+vcy*rmy+vcz*rmz);
    float dmin = dmin_p[0], dmax = dmax_p[0];
    float smag = (nrm - dmin) / (dmax - dmin + 1e-8f);
    edn = fabsf(smag) * (nrm / (nrm + 1e-8f));
    ea = edge_attr[e];
  }

  // ---- edge_enc: [edn, ea] -> edge_latent (inter cols 256..383) ----
  stage_w128(wlds, P.ee1, tid);
  for (int i = lane; i < 512; i += 32) inb[i] = (f16)0.f;
  lds_fence();
  if (lane < 16) { inb[lane*32 + 0] = (f16)edn; inb[lane*32 + 1] = (f16)ea; }
  lds_fence();
  gemm_128<1>(inb + M*32, hi, lane, P.ee0, P.ee_b0, true, hid, 128, nullptr, 0);
  lds_fence();
  gemm_128<4>(hid + M*128, hi, lane, wlds, P.ee_b1, false, nullptr, 0, scr, 132);
  lds_fence();
  ln_rows(scr, 132, P.ee_g, P.ee_be, inter + 256, 384, 0, lane);
  lds_fence();

  // ---- edge_feat(sf) -> sl (inter cols 0..127) ----
  stage_w128(wlds, P.ef1, tid);
  if (lane < 16) for (int j = 0; j < 6; ++j) inb[lane*32 + j] = (f16)sfv[j];
  lds_fence();
  gemm_128<1>(inb + M*32, hi, lane, P.ef0, P.ef_b0, true, hid, 128, nullptr, 0);
  lds_fence();
  gemm_128<4>(hid + M*128, hi, lane, wlds, P.ef_b1, false, nullptr, 0, scr, 132);
  lds_fence();
  ln_rows(scr, 132, P.ef_g, P.ef_be, inter + 0, 384, 0, lane);
  lds_fence();

  // ---- edge_feat(rf) -> rl, accumulate into sl (wlds still holds ef1) ----
  if (lane < 16) for (int j = 0; j < 6; ++j) inb[lane*32 + j] = (f16)rfv[j];
  lds_fence();
  gemm_128<1>(inb + M*32, hi, lane, P.ef0, P.ef_b0, true, hid, 128, nullptr, 0);
  lds_fence();
  gemm_128<4>(hid + M*128, hi, lane, wlds, P.ef_b1, false, nullptr, 0, scr, 132);
  lds_fence();
  ln_rows(scr, 132, P.ef_g, P.ef_be, inter + 0, 384, 1, lane);
  lds_fence();

  // ---- stage inter_enc layer-1 weights ----
  stage_w128(wlds, P.ie1, tid);

  // ---- node_sum gather via async-to-LDS (f16 latent rows, L2-resident) ----
  {
    f16* gbuf = (f16*)scr;  // 32 rows x 128 halves = 8KB staging in scr region
    for (int t = 0; t < 16; ++t) {
      int idx = lane + 32 * t;          // 0..511 b128 transfers
      int row = idx >> 4;               // 0..15 sender rows, 16..31 receiver rows
      int seg = idx & 15;               // 16B segment within a 256B row
      int e = min(e0 + (row & 15), nE - 1);
      int node = (row < 16) ? senders[e] : receivers[e];
      async_copy_b128((char*)gbuf + idx * 16,
                      (const char*)(nlat + (size_t)node * 128) + seg * 16);
    }
    async_wait0();
    lds_fence();
    for (int ei = 0; ei < 16; ++ei) {
      const f16* ps = gbuf + ei * 128 + lane * 4;
      const f16* pr = gbuf + 2048 + ei * 128 + lane * 4;
      f16* d = inter + ei * 384 + 128 + lane * 4;
      for (int j = 0; j < 4; ++j) d[j] = (f16)((float)ps[j] + (float)pr[j]);
    }
    lds_fence();
  }

  // ---- inter_enc: [sl+rl | node_sum | edge_latent] (384) -> il ----
  gemm_128<12>(inter + M*384, hi, lane, P.ie0, P.ie_b0, true, hid, 128, nullptr, 0);
  lds_fence();
  gemm_128<4>(hid + M*128, hi, lane, wlds, P.ie_b1, false, nullptr, 0, scr, 132);
  lds_fence();
  ln_rows(scr, 132, P.ie_g, P.ie_be, hid, 128, 0, lane);
  lds_fence();

  if (!use_hist) {
    int e = e0 + M;
    if (e < nE) {
      f16* hp = hist + (size_t)e * 128 + hi * 64;
      const f16* sp = hid + M * 128 + hi * 64;
#pragma unroll
      for (int c = 0; c < 8; ++c) *(v8h*)(hp + c * 8) = *(const v8h*)(sp + c * 8);
    }
  } else {
    int e = min(e0 + M, nE - 1);
    const f16* hp = hist + (size_t)e * 128 + hi * 64;
    const f16* sp = hid + M * 128 + hi * 64;
    float* dr = scr + M * 132 + hi * 64;
    for (int i = 0; i < 64; ++i) dr[i] = (float)sp[i] + (float)hp[i];
    lds_fence();
    ln_rows(scr, 132, P.lng, P.lnb, hid, 128, 0, lane);
  }
  lds_fence();

  // ---- i1_dec -> coeffs -> fij scatter ----
  stage_w128(wlds, P.i10, tid);
  gemm_128<4>(hid + M*128, hi, lane, wlds, P.i1_b0, true, inter, 384, nullptr, 0); // reuse cols 0..127
  lds_fence();
  gemm_16o<4>(inter + M*384, hi, lane, P.i11, P.i1_b1, 3, scr, 16);
  lds_fence();
  if (lane < 16 && valid) {
    float c0 = scr[lane*16 + 0], c1 = scr[lane*16 + 1], c2 = scr[lane*16 + 2];
    float fx = c0*vax + c1*vbx + c2*vcx;
    float fy = c0*vay + c1*vby + c2*vcy;
    float fz = c0*vaz + c1*vbz + c2*vcz;
    atomicAdd(&out_fij[recv*3 + 0], fx);
    atomicAdd(&out_fij[recv*3 + 1], fy);
    atomicAdd(&out_fij[recv*3 + 2], fz);
  }
}

// ---------------- node residual: m/fext decoders + Ri + G reduction ----------------
#define NR_WAVE_BYTES (16*128*2 + 16*132*4)
__global__ void node_resid_kernel(const f16* m0, const f16* m1, const float* m_b0, const float* m_b1,
                                  const f16* f0, const f16* f1, const float* f_b0, const float* f_b1,
                                  const f16* nlat, const float* a1, const float* fij,
                                  float* Ri, float* totres, float* G, float gd, int N) {
  extern __shared__ char sm[];
  const int lane = threadIdx.x & 31, wave = threadIdx.x >> 5;
  char* wb = sm + (size_t)wave * NR_WAVE_BYTES;
  f16*  hid = (f16*)wb;
  float* scr = (float*)(wb + 16*128*2);
  const int hi = lane >> 4, M = lane & 15;
  const int base = (blockIdx.x * 4 + wave) * 16;
  const f16* aRow = nlat + (size_t)min(base + M, N - 1) * 128;

  gemm_128<4>(aRow, hi, lane, m0, m_b0, true, hid, 128, nullptr, 0);
  lds_fence();
  gemm_16o<4>(hid + M*128, hi, lane, m1, m_b1, 1, scr, 16);
  lds_fence();
  float mval = (lane < 16) ? scr[lane * 16] : 0.f;
  lds_fence();
  gemm_128<4>(aRow, hi, lane, f0, f_b0, true, hid, 128, nullptr, 0);
  lds_fence();
  gemm_16o<4>(hid + M*128, hi, lane, f1, f_b1, 3, scr, 16);
  lds_fence();
  if (lane < 16) {
    int n = base + lane;
    if (n < N) {
      float rx = mval * a1[n*3+0] + fij[n*3+0] - scr[lane*16+0];
      float ry = mval * a1[n*3+1] + fij[n*3+1] - scr[lane*16+1];
      float rz = mval * a1[n*3+2] + fij[n*3+2] - scr[lane*16+2];
      Ri[n*3+0] = rx; Ri[n*3+1] = ry; Ri[n*3+2] = rz;
      totres[n] += gd * sqrtf(rx*rx + ry*ry + rz*rz);
      float v3[3] = {rx, ry, rz};
      for (int j = 0; j < 3; ++j)
        for (int k = 0; k < 3; ++k) atomicAdd(&G[j*3 + k], v3[j] * v3[k]);
    }
  }
}

// ---------------- corrector MLP + Newmark re-integration ----------------
#define CO_WAVE_BYTES (16*32*2 + 16*128*2 + 16*132*4)
__global__ void corrector_kernel(const f16* c0w, const f16* c1w, const float* c_b0, const float* c_b1,
                                 const float* Ri, const float* Gbuf,
                                 float* x1, float* a1, float* v1,
                                 const float* pos, const float* vel, const float* acc,
                                 float gd, int N) {
  extern __shared__ char sm[];
  const int lane = threadIdx.x & 31, wave = threadIdx.x >> 5;
  char* wb = sm + (size_t)wave * CO_WAVE_BYTES;
  f16*  inb = (f16*)wb;
  f16*  hid = (f16*)(wb + 16*32*2);
  float* scr = (float*)(wb + 16*32*2 + 16*128*2);
  const int hi = lane >> 4, M = lane & 15;
  const int base = (blockIdx.x * 4 + wave) * 16;

  float rx=0, ry=0, rz=0, gx=0, gy=0, gz=0;
  for (int i = lane; i < 512; i += 32) inb[i] = (f16)0.f;
  lds_fence();
  if (lane < 16) {
    int ncl = min(base + lane, N - 1);
    rx = Ri[ncl*3+0]; ry = Ri[ncl*3+1]; rz = Ri[ncl*3+2];
    float inv = 1.f / (float)(N + 1);
    float Gm[9];
    for (int t = 0; t < 9; ++t) Gm[t] = Gbuf[t] * inv;
    gx = rx*Gm[0] + ry*Gm[1] + rz*Gm[2];
    gy = rx*Gm[3] + ry*Gm[4] + rz*Gm[5];
    gz = rx*Gm[6] + ry*Gm[7] + rz*Gm[8];
    float nr = sqrtf(rx*rx + ry*ry + rz*rz);
    float ng = sqrtf(gx*gx + gy*gy + gz*gz);
    inb[lane*32 + 0] = (f16)nr;
    inb[lane*32 + 1] = (f16)ng;
    inb[lane*32 + 2] = (f16)(gd * nr);
  }
  lds_fence();
  gemm_128<1>(inb + M*32, hi, lane, c0w, c_b0, true, hid, 128, nullptr, 0);
  lds_fence();
  gemm_16o<4>(hid + M*128, hi, lane, c1w, c_b1, 2, scr, 16);
  lds_fence();
  if (lane < 16) {
    int n = base + lane;
    if (n < N) {
      float ca = scr[lane*16 + 0], cb = scr[lane*16 + 1];
      float nx3[3] = {x1[n*3+0] - (ca*rx + cb*gx),
                      x1[n*3+1] - (ca*ry + cb*gy),
                      x1[n*3+2] - (ca*rz + cb*gz)};
      for (int j = 0; j < 3; ++j) {
        int i = n*3 + j;
        float ac = acc[i];
        float aa = (nx3[j] - pos[i] - DT_C * vel[i]) / (BETA_C * DT_C * DT_C)
                   - (1.f / (2.f * BETA_C) - 1.f) * ac;
        x1[i] = nx3[j];
        a1[i] = aa;
        v1[i] = vel[i] + DT_C * ((1.f - GAM_C) * ac + GAM_C * aa);
      }
    }
  }
}

// ---------------- host side ----------------
extern "C" void kernel_launch(void* const* d_in, const int* in_sizes, int n_in,
                              void* d_out, int out_size, void* d_ws, size_t ws_size,
                              hipStream_t stream) {
  const int N = in_sizes[0] / 3;   // 30000
  const int E = in_sizes[5] / 2;   // 480000
  const float* pos   = (const float*)d_in[0];
  const float* vel   = (const float*)d_in[1];
  const float* pvel  = (const float*)d_in[2];
  const float* eattr = (const float*)d_in[3];
  const int*   ntype = (const int*)d_in[4];
  const int*   eidx  = (const int*)d_in[5];
  const float* vs_p  = (const float*)d_in[6];
  const float* dmn_p = (const float*)d_in[7];
  const float* dmx_p = (const float*)d_in[8];
  const int* senders = eidx;
  const int* receivers = eidx + E;
  auto fp = [&](int i) { return (const float*)d_in[i]; };

  char* base = (char*)d_ws;
  size_t off = 0;
  auto alloc = [&](size_t bytes) -> char* {
    char* p = base + off;
    off = (off + bytes + 255) & ~(size_t)255;
    return p;
  };
  float* acc    = (float*)alloc((size_t)N * 3 * 4);
  float* x1     = (float*)alloc((size_t)N * 3 * 4);
  float* a1     = (float*)alloc((size_t)N * 3 * 4);
  float* v1     = (float*)alloc((size_t)N * 3 * 4);
  float* totres = (float*)alloc((size_t)N * 4);
  float* Ri     = (float*)alloc((size_t)N * 3 * 4);
  float* fij    = (float*)alloc((size_t)N * 3 * 4);
  float* G      = (float*)alloc(16 * 4);
  f16*   nlat   = (f16*)alloc((size_t)N * 128 * 2);
  f16*   hist   = (f16*)alloc((size_t)E * 128 * 2);

  auto conv = [&](int idx, int K, int Nw, int Kp, int Np) -> f16* {
    f16* d = (f16*)alloc((size_t)Kp * Np * 2);
    int tot = Kp * Np;
    conv_w_kernel<<<dim3((tot + 255) / 256), dim3(256), 0, stream>>>(fp(idx), d, K, Nw, Kp, Np);
    return d;
  };

  // param leaf indices (jax pytree order: dict keys sorted; 'bs' < 'ln' < 'ws')
  struct BW { f16 *ee0,*ee1,*ef0,*ef1,*fx0,*fx1,*i10,*i11,*ie0,*ie1,*m0,*m1; };
  BW W[2];
  for (int b = 0; b < 2; ++b) {
    int bb = 9 + 32 * b;
    W[b].ee0 = conv(bb + 4,  2,   128, 32,  128);
    W[b].ee1 = conv(bb + 5,  128, 128, 128, 128);
    W[b].ef0 = conv(bb + 10, 6,   128, 32,  128);
    W[b].ef1 = conv(bb + 11, 128, 128, 128, 128);
    W[b].fx0 = conv(bb + 14, 128, 128, 128, 128);
    W[b].fx1 = conv(bb + 15, 128, 3,   128, 16);
    W[b].i10 = conv(bb + 18, 128, 128, 128, 128);
    W[b].i11 = conv(bb + 19, 128, 3,   128, 16);
    W[b].ie0 = conv(bb + 24, 384, 128, 384, 128);
    W[b].ie1 = conv(bb + 25, 128, 128, 128, 128);
    W[b].m0  = conv(bb + 30, 128, 128, 128, 128);
    W[b].m1  = conv(bb + 31, 128, 1,   128, 16);
  }
  f16* cw0 = conv(75, 3,   128, 32,  128);
  f16* cw1 = conv(76, 128, 2,   128, 16);
  f16* nw0 = conv(81, 2,   128, 32,  128);
  f16* nw1 = conv(82, 128, 128, 128, 128);

  prep_kernel<<<dim3((N + 255) / 256), dim3(256), 0, stream>>>(pos, vel, pvel, acc, x1, a1, v1, totres, N);

  int gN = (N + 63) / 64;
  node_enc_kernel<<<dim3(gN), dim3(128), 4 * NE_WAVE_BYTES, stream>>>(
      nw0, nw1, fp(77), fp(78), fp(79), fp(80), ntype, vel, nlat, N);

  int gE = (E + 63) / 64;
  for (int b = 0; b < 2; ++b) {
    int bb = 9 + 32 * b;
    float gd = (b == 0) ? 0.8f : 1.0f;

    zero_kernel<<<dim3((N * 3 + 255) / 256), dim3(256), 0, stream>>>(fij, N * 3);
    zero_kernel<<<dim3(1), dim3(256), 0, stream>>>(G, 16);

    EdgeParams P;
    P.ee0 = W[b].ee0; P.ee1 = W[b].ee1; P.ef0 = W[b].ef0; P.ef1 = W[b].ef1;
    P.ie0 = W[b].ie0; P.ie1 = W[b].ie1; P.i10 = W[b].i10; P.i11 = W[b].i11;
    P.ee_b0 = fp(bb + 0);  P.ee_b1 = fp(bb + 1);  P.ee_g = fp(bb + 2);  P.ee_be = fp(bb + 3);
    P.ef_b0 = fp(bb + 6);  P.ef_b1 = fp(bb + 7);  P.ef_g = fp(bb + 8);  P.ef_be = fp(bb + 9);
    P.ie_b0 = fp(bb + 20); P.ie_b1 = fp(bb + 21); P.ie_g = fp(bb + 22); P.ie_be = fp(bb + 23);
    P.i1_b0 = fp(bb + 16); P.i1_b1 = fp(bb + 17);
    P.lnb = fp(bb + 26);   P.lng = fp(bb + 27);

    edge_block_kernel<<<dim3(gE), dim3(128), EDGE_SMEM, stream>>>(
        P, senders, receivers, eattr, x1, v1, vel, vs_p, dmn_p, dmx_p,
        nlat, hist, (b == 0) ? 0 : 1, fij, E);

    node_resid_kernel<<<dim3(gN), dim3(128), 4 * NR_WAVE_BYTES, stream>>>(
        W[b].m0, W[b].m1, fp(bb + 28), fp(bb + 29),
        W[b].fx0, W[b].fx1, fp(bb + 12), fp(bb + 13),
        nlat, a1, fij, Ri, totres, G, gd, N);

    corrector_kernel<<<dim3(gN), dim3(128), 4 * CO_WAVE_BYTES, stream>>>(
        cw0, cw1, fp(73), fp(74), Ri, G, x1, a1, v1, pos, vel, acc, gd, N);
  }

  final_kernel<<<dim3((N + 255) / 256), dim3(256), 0, stream>>>(x1, v1, totres, (float*)d_out, N);
  (void)n_in; (void)in_sizes; (void)out_size; (void)ws_size;
}
